// MultiQueryAttention_36756330119711
// MI455X (gfx1250) — matrix operand order repaired and checked by reading, not compiled, not run on hardware
//
#include <hip/hip_runtime.h>
#include <hip/hip_bf16.h>
#include <stdint.h>
#include <math.h>

// ---------------------------------------------------------------------------
// MQA: x[B,S,1024] -> Q(16 heads) / K,V(1 shared head, Dk=64) -> flash attn
//      -> O-proj.  GEMMs/attention use v_wmma_f32_16x16x32_f16 (f16 in,
//      f32 acc). K/V tiles staged into LDS with the Tensor Data Mover
//      (tensor_load_to_lds + s_wait_tensorcnt), with TDM pad = LDS stride 72.
// ---------------------------------------------------------------------------

#define DM   1024
#define HN   16
#define DK   64
#define BATCH 2
#define SEQ  2048
#define MTOT (BATCH*SEQ)       // 4096 rows

typedef _Float16 v16h __attribute__((ext_vector_type(16)));
typedef _Float16 h8   __attribute__((ext_vector_type(8)));
typedef float    v8f  __attribute__((ext_vector_type(8)));
typedef unsigned int u32x4 __attribute__((ext_vector_type(4)));
typedef int          i32x8 __attribute__((ext_vector_type(8)));
typedef int          i32x4 __attribute__((ext_vector_type(4)));

union HFrag { v16h h; uint32_t u[8]; };

__device__ __forceinline__ uint32_t pack2h(float lo, float hi) {
  union { _Float16 h[2]; uint32_t u; } t;
  t.h[0] = (_Float16)lo; t.h[1] = (_Float16)hi;
  return t.u;
}

__device__ __forceinline__ v16h mk16(h8 lo, h8 hi) {
  v16h r;
#pragma unroll
  for (int i = 0; i < 8; i++) { r[i] = lo[i]; r[i + 8] = hi[i]; }
  return r;
}

// TDM: load a 64x64 f16 tile (row-major, row pitch = stride_d0 elements) from
// a 2-D tensor [tensor_d1][tensor_d0] into LDS at lds_off, padding each
// 128-byte row with 16 bytes -> LDS row stride 72 halfwords (144 B).
__device__ __forceinline__ void tdm_load_tile64(uint32_t lds_off,
                                                const void* tile_start,
                                                uint32_t tensor_d0,
                                                uint32_t tensor_d1,
                                                uint32_t stride_d0) {
  const uint64_t ga = (uint64_t)(uintptr_t)tile_start;
  u32x4 g0;
  g0[0] = 1u;                                   // count=1 valid descriptor
  g0[1] = lds_off;                              // D#.lds_addr
  g0[2] = (uint32_t)ga;                         // global_addr[31:0]
  g0[3] = (uint32_t)(ga >> 32) | (2u << 30);    // global_addr[56:32] | type=2
  i32x8 g1;
  g1[0] = (1 << 16)      // data_size = 2 bytes
        | (1 << 20)      // pad_enable
        | (4 << 22)      // pad_interval: 32 DWORDs (one 128B tile row)
        | (3 << 25);     // pad_amount: 4 DWORDs (8 f16) -> stride 72 halves
  g1[1] = (int)((tensor_d0 & 0xFFFFu) << 16);                       // dim0 lo16
  g1[2] = (int)((tensor_d0 >> 16) | ((tensor_d1 & 0xFFFFu) << 16)); // dim0 hi | dim1 lo
  g1[3] = (int)((tensor_d1 >> 16) | (64u << 16));                   // dim1 hi | tile_dim0=64
  g1[4] = 64;            // tile_dim1 = 64, tile_dim2 = 0
  g1[5] = (int)stride_d0;// tensor_dim0_stride lo32 (elements)
  g1[6] = 0;             // stride hi16 | dim1_stride lo16 (unused for 2D tile)
  g1[7] = 0;
  const i32x4 z4 = {0, 0, 0, 0};
  const i32x8 z8 = {0, 0, 0, 0, 0, 0, 0, 0};
  // amdgpu-toolchain (clang-23) 6-arg form: (g0, g1, g2, g3, extra, cpol).
  __builtin_amdgcn_tensor_load_to_lds(g0, g1, z4, z4, z8, 0);
}

// --------------------------- conversion kernels ----------------------------

__global__ void cvt_f32_to_f16(const float* __restrict__ src,
                               _Float16* __restrict__ dst, int n) {
  int i = (blockIdx.x * blockDim.x + threadIdx.x) * 4;
  if (i + 3 < n) {
    float4 v = *(const float4*)(src + i);
    dst[i]     = (_Float16)v.x;
    dst[i + 1] = (_Float16)v.y;
    dst[i + 2] = (_Float16)v.z;
    dst[i + 3] = (_Float16)v.w;
  }
}

// W[K][N] (fp32, row-major) -> Wt[N][K] (f16): B-fragments become contiguous.
__global__ void cvt_w_transpose(const float* __restrict__ W,
                                _Float16* __restrict__ Wt, int K, int N) {
  int idx = blockIdx.x * blockDim.x + threadIdx.x;
  if (idx < K * N) {
    int n = idx / K;
    int k = idx - n * K;
    Wt[idx] = (_Float16)W[(size_t)k * N + n];
  }
}

// ------------------------------- WMMA GEMM ---------------------------------
// C = (A[M][K] @ Wt[N][K]^T + bias) * scale.
// 128 threads = 4 waves; each wave owns a 32(M) x 64(N) tile (8 accumulators),
// K-loop step 32: 2 A-frags + 4 B-frags -> 8 WMMAs (24 B of operands / WMMA).
// Whole working set (~30 MB) lives in the 192 MB L2, so operands stream from
// global at L2 bandwidth; no LDS staging needed.
// OMODE: 0 = f16 C[m][n], 1 = f32 C[m][n], 2 = f16 transposed C[n][m].

template <int OMODE>
__global__ void __launch_bounds__(128)
gemm_wmma_f16(const _Float16* __restrict__ A,   // [M][K] f16
              const _Float16* __restrict__ Wt,  // [N][K] f16
              const float* __restrict__ bias,   // [N]
              void* __restrict__ Cout,
              int M, int N, int K, float scale) {
  const int lane = threadIdx.x & 31;
  const int wave = threadIdx.x >> 5;
  const int lg = lane >> 4;       // lane-group (0/1)
  const int ln = lane & 15;
  const int m0 = (blockIdx.x * 4 + wave) * 32;
  const int n0 = blockIdx.y * 64;

  v8f acc[2][4] = {};
  const _Float16* arow0 = A + (size_t)(m0 + ln) * K;
  const _Float16* arow1 = A + (size_t)(m0 + 16 + ln) * K;

  for (int k0 = 0; k0 < K; k0 += 32) {
    // A frag (16x32): lane row = m; v0-3: k = k0+8*lg+0..7, v4-7: +16.
    v16h af0 = mk16(*(const h8*)(arow0 + k0 + 8 * lg),
                    *(const h8*)(arow0 + k0 + 16 + 8 * lg));
    v16h af1 = mk16(*(const h8*)(arow1 + k0 + 8 * lg),
                    *(const h8*)(arow1 + k0 + 16 + 8 * lg));
#pragma unroll
    for (int nt = 0; nt < 4; nt++) {
      // B frag (32x16): lane col = n; v0-3: k = k0+8*lg+0..7, v4-7: +16.
      const _Float16* wrow =
          Wt + (size_t)(n0 + nt * 16 + ln) * K + k0 + 8 * lg;
      v16h bf = mk16(*(const h8*)wrow, *(const h8*)(wrow + 16));
      acc[0][nt] = __builtin_amdgcn_wmma_f32_16x16x32_f16(
          false, af0, false, bf, (short)0, acc[0][nt], false, false);
      acc[1][nt] = __builtin_amdgcn_wmma_f32_16x16x32_f16(
          false, af1, false, bf, (short)0, acc[1][nt], false, false);
    }
  }

#pragma unroll
  for (int mt = 0; mt < 2; mt++) {
#pragma unroll
    for (int nt = 0; nt < 4; nt++) {
      const int n = n0 + nt * 16 + ln;
      const float bv = bias[n];
#pragma unroll
      for (int r = 0; r < 8; r++) {
        const int m = m0 + mt * 16 + r + 8 * lg;  // C: M = vgpr + 8*lanegroup
        float v = (acc[mt][nt][r] + bv) * scale;
        if (OMODE == 1)
          ((float*)Cout)[(size_t)m * N + n] = v;
        else if (OMODE == 2)
          ((_Float16*)Cout)[(size_t)n * M + m] = (_Float16)v;   // transposed
        else
          ((_Float16*)Cout)[(size_t)m * N + n] = (_Float16)v;
      }
    }
  }
}

// --------------------------- flash attention -------------------------------
// grid.x = SEQ/64 query tiles, grid.y = B*H. 128 threads = 4 waves, each wave
// owns 16 queries. Computes S^T = K_tile x Q^T, online softmax, O^T = V^T x P^T.
// K and V^T 64x64 tiles are DMA'd into LDS by the Tensor Data Mover with
// hardware row padding (stride 72 halves). P^T (C->B frag) conversion is
// per-lane f16 packing + one shfl_xor(16).

__global__ void __launch_bounds__(128)
mqa_flash_attn(const _Float16* __restrict__ Qh,  // [B*S][1024], pre-scaled 1/8
               const _Float16* __restrict__ Kh,  // [B*S][64]
               const _Float16* __restrict__ VT,  // [64][B*S]  (V transposed)
               _Float16* __restrict__ AO) {      // [B*S][1024]
  __shared__ __align__(16) _Float16 Ks[64][72];  // [key][d]  (144 B rows)
  __shared__ __align__(16) _Float16 Vt[64][72];  // [d][key]

  const int lane = threadIdx.x & 31;
  const int wave = threadIdx.x >> 5;
  const int lg = lane >> 4;
  const int ln = lane & 15;
  const int b = blockIdx.y >> 4, h = blockIdx.y & 15;
  const int q = blockIdx.x * 64 + wave * 16 + ln;   // this lane's query row

  // Q^T B-fragments (N = query = ln, K = d), reused across all key tiles.
  const _Float16* qrow = Qh + ((size_t)(b * SEQ + q)) * DM + h * DK;
  v16h Qb0 = mk16(*(const h8*)(qrow + 8 * lg),
                  *(const h8*)(qrow + 16 + 8 * lg));
  v16h Qb1 = mk16(*(const h8*)(qrow + 32 + 8 * lg),
                  *(const h8*)(qrow + 48 + 8 * lg));

  v8f Oacc[4] = {};                 // O^T: d = dt*16 + r + 8*lg, query = ln
  float m_i = -INFINITY, l_i = 0.0f;

  const _Float16* Kb  = Kh + (size_t)b * SEQ * DK;
  const _Float16* VTb = VT + (size_t)b * SEQ;     // column offset into [64][4096]
  const uint32_t ksOff = (uint32_t)(uintptr_t)&Ks[0][0];
  const uint32_t vtOff = (uint32_t)(uintptr_t)&Vt[0][0];

  for (int kt = 0; kt < SEQ / 64; ++kt) {
    __syncthreads();   // previous iteration done reading LDS
    if (wave == 0) {
      // K tile: rows = 64 keys, row pitch DK;  V^T tile: rows = 64 d, pitch MTOT.
      tdm_load_tile64(ksOff, Kb + (size_t)kt * 64 * DK, DK, SEQ, DK);
      tdm_load_tile64(vtOff, VTb + (size_t)kt * 64, MTOT, DK, MTOT);
      __builtin_amdgcn_s_wait_tensorcnt(0);
    }
    if (kt + 1 < SEQ / 64)  // warm L2 for the next K tile (global_prefetch_b8)
      __builtin_prefetch(Kb + (size_t)(kt + 1) * 64 * DK + threadIdx.x * 64, 0, 1);
    __syncthreads();

    // S^T tiles: 4 x (16 keys x 16 queries), accumulate over d in 2 chunks.
    v8f Sacc[4];
#pragma unroll
    for (int st = 0; st < 4; st++) {
      const _Float16* kr = &Ks[st * 16 + ln][0];  // A frag: M = key = ln
      v16h a0 = mk16(*(const h8*)(kr + 8 * lg),
                     *(const h8*)(kr + 16 + 8 * lg));
      v16h a1 = mk16(*(const h8*)(kr + 32 + 8 * lg),
                     *(const h8*)(kr + 48 + 8 * lg));
      v8f z = {};
      z = __builtin_amdgcn_wmma_f32_16x16x32_f16(false, a0, false, Qb0,
                                                 (short)0, z, false, false);
      z = __builtin_amdgcn_wmma_f32_16x16x32_f16(false, a1, false, Qb1,
                                                 (short)0, z, false, false);
      Sacc[st] = z;
    }

    // Online softmax: query = ln; this lane holds half the keys, partner l^16
    // the other half -> one shfl_xor for the max and one for the sum.
    float tmax = -INFINITY;
#pragma unroll
    for (int st = 0; st < 4; st++)
#pragma unroll
      for (int r = 0; r < 8; r++) tmax = fmaxf(tmax, Sacc[st][r]);
    tmax = fmaxf(tmax, __shfl_xor(tmax, 16));
    const float m_new = fmaxf(m_i, tmax);
    const float corr = __expf(m_i - m_new);   // 0 on first tile (m_i = -inf)
    float psum = 0.0f;
#pragma unroll
    for (int st = 0; st < 4; st++)
#pragma unroll
      for (int r = 0; r < 8; r++) {
        float p = __expf(Sacc[st][r] - m_new);
        Sacc[st][r] = p;
        psum += p;
      }
    psum += __shfl_xor(psum, 16);
    l_i = l_i * corr + psum;
    m_i = m_new;
#pragma unroll
    for (int dt = 0; dt < 4; dt++) Oacc[dt] *= corr;

    // O^T += V^T x P^T, key chunks of 32.
#pragma unroll
    for (int c = 0; c < 2; c++) {
      HFrag pb;  // P^T B-frag: N = query = ln, K = key = 16*lg + elem
#pragma unroll
      for (int j = 0; j < 4; j++) {
        uint32_t pkA = pack2h(Sacc[2 * c][2 * j],     Sacc[2 * c][2 * j + 1]);
        uint32_t pkB = pack2h(Sacc[2 * c + 1][2 * j], Sacc[2 * c + 1][2 * j + 1]);
        uint32_t sent = lg ? pkA : pkB;                  // what partner needs
        uint32_t recv = (uint32_t)__shfl_xor((int)sent, 16);
        pb.u[j]     = lg ? recv : pkA;
        pb.u[4 + j] = lg ? pkB  : recv;
      }
#pragma unroll
      for (int dt = 0; dt < 4; dt++) {
        const _Float16* vr = &Vt[dt * 16 + ln][c * 32];  // A frag: M = d = ln
        v16h vf = mk16(*(const h8*)(vr + 16 * lg),
                       *(const h8*)(vr + 16 * lg + 8));
        Oacc[dt] = __builtin_amdgcn_wmma_f32_16x16x32_f16(
            false, vf, false, pb.h, (short)0, Oacc[dt], false, false);
      }
    }
  }

  // Normalize and store: AO[b*S+q][h*64 + d], d = dt*16 + r + 8*lg.
  const float inv = 1.0f / l_i;
  _Float16* orow = AO + ((size_t)(b * SEQ + q)) * DM + h * DK;
#pragma unroll
  for (int dt = 0; dt < 4; dt++)
#pragma unroll
    for (int r = 0; r < 8; r++)
      orow[dt * 16 + r + 8 * lg] = (_Float16)(Oacc[dt][r] * inv);
}

// ------------------------------- launcher ----------------------------------

extern "C" void kernel_launch(void* const* d_in, const int* in_sizes, int n_in,
                              void* d_out, int out_size, void* d_ws,
                              size_t ws_size, hipStream_t stream) {
  const float* x  = (const float*)d_in[0];
  const float* Wq = (const float*)d_in[1];
  const float* bq = (const float*)d_in[2];
  const float* Wk = (const float*)d_in[3];
  const float* bk = (const float*)d_in[4];
  const float* Wv = (const float*)d_in[5];
  const float* bv = (const float*)d_in[6];
  const float* Wo = (const float*)d_in[7];
  const float* bo = (const float*)d_in[8];

  // Workspace carve-out (~30.7 MB total, 256B aligned chunks).
  size_t off = 0;
  auto carve = [&](size_t bytes) {
    void* p = (char*)d_ws + off;
    off += (bytes + 255) & ~(size_t)255;
    return p;
  };
  _Float16* xh  = (_Float16*)carve((size_t)MTOT * DM * 2);
  _Float16* Wqt = (_Float16*)carve((size_t)DM * DM * 2);
  _Float16* Wkt = (_Float16*)carve((size_t)DM * DK * 2);
  _Float16* Wvt = (_Float16*)carve((size_t)DM * DK * 2);
  _Float16* Wot = (_Float16*)carve((size_t)DM * DM * 2);
  _Float16* Qh  = (_Float16*)carve((size_t)MTOT * DM * 2);
  _Float16* Kh  = (_Float16*)carve((size_t)MTOT * DK * 2);
  _Float16* VT  = (_Float16*)carve((size_t)DK * MTOT * 2);   // [64][4096]
  _Float16* AOh = (_Float16*)carve((size_t)MTOT * DM * 2);

  // 1) fp32 -> f16 conversions (weights also transposed to [N][K]).
  cvt_f32_to_f16<<<(MTOT * DM / 4 + 255) / 256, 256, 0, stream>>>(
      x, xh, MTOT * DM);
  cvt_w_transpose<<<(DM * DM + 255) / 256, 256, 0, stream>>>(Wq, Wqt, DM, DM);
  cvt_w_transpose<<<(DM * DK + 255) / 256, 256, 0, stream>>>(Wk, Wkt, DM, DK);
  cvt_w_transpose<<<(DM * DK + 255) / 256, 256, 0, stream>>>(Wv, Wvt, DM, DK);
  cvt_w_transpose<<<(DM * DM + 255) / 256, 256, 0, stream>>>(Wo, Wot, DM, DM);

  // 2) Projections. 1/sqrt(Dk)=0.125 folded into Q; V written transposed.
  gemm_wmma_f16<0><<<dim3(MTOT / 128, DM / 64), 128, 0, stream>>>(
      xh, Wqt, bq, Qh, MTOT, DM, DM, 0.125f);
  gemm_wmma_f16<0><<<dim3(MTOT / 128, 1), 128, 0, stream>>>(
      xh, Wkt, bk, Kh, MTOT, DK, DM, 1.0f);
  gemm_wmma_f16<2><<<dim3(MTOT / 128, 1), 128, 0, stream>>>(
      xh, Wvt, bv, VT, MTOT, DK, DM, 1.0f);

  // 3) Flash attention (scores never hit memory; TDM feeds LDS).
  mqa_flash_attn<<<dim3(SEQ / 64, BATCH * HN), 128, 0, stream>>>(
      Qh, Kh, VT, AOh);

  // 4) Output projection, fp32 result.
  gemm_wmma_f16<1><<<dim3(MTOT / 128, DM / 64), 128, 0, stream>>>(
      AOh, Wot, bo, (float*)d_out, MTOT, DM, DM, 1.0f);
}
